// LierePositionEncoder_42288247996733
// MI455X (gfx1250) — matrix operand order repaired and checked by reading, not compiled, not run on hardware
//
#include <hip/hip_runtime.h>

// CDNA5 (gfx1250) LIERE position encoder:
//   out[n,l,g] = expm(x_n*B0[l,g] + y_n*B1[l,g]),  B = skew(gen_params) 8x8.
// 524,800 independent 8x8 expm's, 134 MB f32 out (~5.8us at 23.3 TB/s HBM).
// Compute-bound on small matmuls -> route them through V_WMMA_F32_16X16X4_F32
// by packing two 8x8 matrices block-diagonally per 16x16 wave tile.
// expm via scaling (2^-8) + degree-8 Taylor (Horner) + 8 squarings; uniform
// control flow (WMMA needs EXEC all-ones). Fragment marshalling goes through
// a per-wave stride-17-padded LDS tile (conflict-free on 64 banks); a wave's
// DS ops are in-order so only compiler scheduling barriers are required.

typedef __attribute__((ext_vector_type(2)))  float    v2f;
typedef __attribute__((ext_vector_type(8)))  float    v8f;
typedef __attribute__((ext_vector_type(16))) _Float16 v16h;

#define NTOK 1025   // 1024 patches + cls (identity rotation)
#define NLG  512    // 4 layers * 128 generators
#define DEG  8      // Taylor degree for exp(A/2^SSQ)
#define SSQ  8      // number of squarings (scale = 2^-8)
#define ROWS 17     // padded LDS row stride in dwords (bank-conflict free)

#if __has_builtin(__builtin_amdgcn_wmma_f32_16x16x4_f32)
#define USE_F32_WMMA 1
#endif

// 16x16 @ 16x16 matmul of two LDS tiles, result in C/D-layout registers.
__device__ __forceinline__ v8f mm16(const float* tA, const float* tB,
                                    const int lcol, const int lhalf) {
  v8f acc = {0.f, 0.f, 0.f, 0.f, 0.f, 0.f, 0.f, 0.f};
#ifdef USE_F32_WMMA
  // four K=4 chunks accumulated through C:
  // A-frag: lanes 0-15 hold (M=lane, K=k0..k0+1), lanes 16-31 (M, K=k0+2..3)
  // B-frag: VGPR v holds row K=k0+2*lhalf+v, col N=lane%16
#pragma unroll
  for (int c = 0; c < 4; ++c) {
    const int kb = c * 4 + 2 * lhalf;
    v2f a, b;
    a.x = tA[lcol * ROWS + kb];
    a.y = tA[lcol * ROWS + kb + 1];
    b.x = tB[kb * ROWS + lcol];
    b.y = tB[(kb + 1) * ROWS + lcol];
    acc = __builtin_amdgcn_wmma_f32_16x16x4_f32(false, a, false, b,
                                                (short)0, acc, false, false);
  }
#else
  // fallback: one 16x16x32 f16 WMMA with K zero-padded to 32
  v16h a, b;
#pragma unroll
  for (int ei = 0; ei < 16; ++ei) {
    const int ka = (ei < 8) ? (ei + 8 * lhalf) : (ei + 8 + 8 * lhalf);
    a[ei] = (ka < 16) ? (_Float16)tA[lcol * ROWS + ka] : (_Float16)0.f;
    const int kb = (ei < 8) ? (ei + 8 * lhalf) : (ei + 8 + 8 * lhalf);
    b[ei] = (kb < 16) ? (_Float16)tB[kb * ROWS + lcol] : (_Float16)0.f;
  }
  acc = __builtin_amdgcn_wmma_f32_16x16x32_f16(false, a, false, b,
                                               (short)0, acc, false, false);
#endif
  return acc;
}

__global__ __launch_bounds__(256) void liere_expm_wmma(
    const float* __restrict__ pos,   // [1024][2]
    const float* __restrict__ gp,    // [2][512][64]
    float* __restrict__ out) {       // [1025][512][64]
  __shared__ float sB[2][64];             // skew bases for this (l,g)
  __shared__ float sW[8][2][16 * ROWS];   // per-wave {A_s tile, T tile}

  const int lg    = blockIdx.x;           // generator index l*128+g
  const int tchk  = blockIdx.y;           // 16-token chunk
  const int tid   = threadIdx.x;
  const int lane  = tid & 31;
  const int wid   = tid >> 5;
  const int lhalf = lane >> 4;            // 0: lanes 0-15, 1: lanes 16-31
  const int lcol  = lane & 15;

  // skew-symmetrize gen_params once per block: B[i,j]=gp[i,j] (i<j),
  // -gp[j,i] (i>j), 0 on diag  (the -0.5 shift cancels exactly).
  if (tid < 128) {
    const int d = tid >> 6, e = tid & 63, i = e >> 3, j = e & 7;
    const float* g0 = gp + ((size_t)d * NLG + lg) * 64;
    float v = 0.f;
    if (i < j)      v =  g0[i * 8 + j];
    else if (i > j) v = -g0[j * 8 + i];
    sB[d][e] = v;
  }
  __syncthreads();

  const int na = tchk * 16 + wid * 2;     // wave owns tokens na, na+1
  if (na >= NTOK) return;                 // wave-uniform exit
  const int nb  = na + 1;
  const int nbc = (nb < NTOK) ? nb : 0;   // clamp: token 0 == cls == zero pos

  float xa = 0.f, ya = 0.f, xb = 0.f, yb = 0.f;
  if (na > 0)  { xa = pos[(na  - 1) * 2 + 0]; ya = pos[(na  - 1) * 2 + 1]; }
  if (nbc > 0) { xb = pos[(nbc - 1) * 2 + 0]; yb = pos[(nbc - 1) * 2 + 1]; }

  float* tA = sW[wid][0];
  float* tT = sW[wid][1];
  const float sc = 1.0f / 256.0f;         // 2^-SSQ

  // scaled block-diagonal A_s into tA (each lane fills its 8 C-layout slots)
#pragma unroll
  for (int r = 0; r < 8; ++r) {
    const int M = r + 8 * lhalf, N = lcol;
    float v = 0.f;
    if (M < 8 && N < 8)
      v = (xa * sB[0][M * 8 + N] + ya * sB[1][M * 8 + N]) * sc;
    else if (M >= 8 && N >= 8)
      v = (xb * sB[0][(M - 8) * 8 + N - 8] + yb * sB[1][(M - 8) * 8 + N - 8]) * sc;
    tA[M * ROWS + N] = v;
  }
  __builtin_amdgcn_wave_barrier();

  // T = I + A_s/DEG   (C/D layout in registers)
  float t[8];
#pragma unroll
  for (int r = 0; r < 8; ++r) {
    const int M = r + 8 * lhalf;
    t[r] = tA[M * ROWS + lcol] * (1.0f / DEG) + ((M == lcol) ? 1.f : 0.f);
  }

  // Horner: T <- I + (A_s/k) * T, k = DEG-1 .. 1
#pragma unroll
  for (int k = DEG - 1; k >= 1; --k) {
    __builtin_amdgcn_wave_barrier();
#pragma unroll
    for (int r = 0; r < 8; ++r) tT[(r + 8 * lhalf) * ROWS + lcol] = t[r];
    __builtin_amdgcn_wave_barrier();
    const v8f m = mm16(tA, tT, lcol, lhalf);
    const float ik = 1.0f / (float)k;
#pragma unroll
    for (int r = 0; r < 8; ++r)
      t[r] = m[r] * ik + (((r + 8 * lhalf) == lcol) ? 1.f : 0.f);
  }

  // 8 squarings: T <- T*T
#pragma unroll
  for (int s = 0; s < SSQ; ++s) {
    __builtin_amdgcn_wave_barrier();
#pragma unroll
    for (int r = 0; r < 8; ++r) tT[(r + 8 * lhalf) * ROWS + lcol] = t[r];
    __builtin_amdgcn_wave_barrier();
    const v8f m = mm16(tT, tT, lcol, lhalf);
#pragma unroll
    for (int r = 0; r < 8; ++r) t[r] = m[r];
  }

  // publish final tile, then coalesced float4-per-lane global stores
  __builtin_amdgcn_wave_barrier();
#pragma unroll
  for (int r = 0; r < 8; ++r) tT[(r + 8 * lhalf) * ROWS + lcol] = t[r];
  __builtin_amdgcn_wave_barrier();

  const int f0  = lane * 4;               // lanes 0-15 -> token a, 16-31 -> b
  const int tok = f0 >> 6;
  const int n   = tok ? nb : na;
  if (n < NTOK) {
    float4 o;
#pragma unroll
    for (int p = 0; p < 4; ++p) {
      const int e = (f0 + p) & 63, i = e >> 3, j = e & 7;
      (&o.x)[p] = tT[(i + 8 * tok) * ROWS + (j + 8 * tok)];
    }
    *(float4*)(out + ((size_t)n * NLG + lg) * 64 + (f0 & 63)) = o;
  }
}

extern "C" void kernel_launch(void* const* d_in, const int* in_sizes, int n_in,
                              void* d_out, int out_size, void* d_ws, size_t ws_size,
                              hipStream_t stream) {
  const float* pos = (const float*)d_in[0];   // [1,1024,2] f32
  const float* gp  = (const float*)d_in[1];   // [2,4,128,8,8] f32
  float* out = (float*)d_out;                 // [1,1025,4,128,8,8] f32

  dim3 grid(NLG, (NTOK + 15) / 16);           // 512 x 65 blocks
  liere_expm_wmma<<<grid, dim3(256), 0, stream>>>(pos, gp, out);
}
